// BondUpdate_1855425871938
// MI455X (gfx1250) — compile-verified, matrix operand order checked by reading, not run on hardware
//
#include <hip/hip_runtime.h>

// ---------------------------------------------------------------------------
// BondUpdate MLP for MI455X (gfx1250), wave32 + WMMA bf16.
//   rows = B*E = 800,000 ; per row: concat(224) -> 128 relu -> 128 relu -> 64
// Block = 256 threads = 8 wave32s; each block processes M=32 rows/iter.
// Each wave owns one 16-wide N-tile, runs two independent accumulator chains
// (two M-tiles) reusing each B fragment. Gather is software-pipelined: next
// tile's global loads are issued before the current tile's WMMA compute.
// ---------------------------------------------------------------------------

typedef __attribute__((ext_vector_type(16))) __bf16 v16bf;
typedef __attribute__((ext_vector_type(4)))  __bf16 v4bf;
typedef __attribute__((ext_vector_type(8)))  float  v8f;

constexpr int SITE_LEN = 64;
constexpr int IN_DIM   = 224;   // 64 + 64 + 64 + 32
constexpr int HD       = 128;
constexpr int OUT_DIM  = 64;
constexpr int BATCH    = 8;
constexpr int NSITES   = 20000;
constexpr int NE       = 100000;
constexpr int ROWS     = BATCH * NE;      // 800000
constexpr int TILE_M   = 32;              // rows per block-iteration
constexpr int NTILES   = ROWS / TILE_M;   // 25000 (exact; NE % 32 == 0 so a
                                          // tile never crosses a batch bound)

constexpr int L1K = IN_DIM / 32;          // 7 K-chunks
constexpr int L2K = HD / 32;              // 4
constexpr int L3K = HD / 32;              // 4
constexpr int L1N = HD / 16;              // 8 N-tiles
constexpr int L2N = HD / 16;              // 8
constexpr int L3N = OUT_DIM / 16;         // 4

constexpr int FRAGS_L1     = L1K * L1N;               // 56
constexpr int FRAGS_L2     = L2K * L2N;               // 32
constexpr int FRAGS_L3     = L3K * L3N;               // 16
constexpr int FRAG_BASE_L2 = FRAGS_L1;                // 56
constexpr int FRAG_BASE_L3 = FRAGS_L1 + FRAGS_L2;     // 88
constexpr int TOT_FRAGS    = FRAG_BASE_L3 + FRAGS_L3; // 104
constexpr int FRAG_ELEMS   = 512;                     // 32 lanes * 16 bf16
constexpr int WPACK_ELEMS  = TOT_FRAGS * FRAG_ELEMS;  // 53248 bf16 = 104 KB

// LDS element offsets (bf16 units)
constexpr int LDS_A  = WPACK_ELEMS;                 // A panel 32x224
constexpr int LDS_H1 = LDS_A + TILE_M * IN_DIM;     // H1 panel 32x128
constexpr int LDS_H2 = LDS_H1 + TILE_M * HD;        // H2 panel 32x128
constexpr int LDS_TOT_ELEMS = LDS_H2 + TILE_M * HD; // 68608 -> 137216 bytes

union Frag {
    v16bf v;
    uint4 q[2];
};

__device__ __forceinline__ v8f wmma_bf16(v16bf a, v16bf b, v8f c) {
    // (neg_a, A, neg_b, B, c_mod, C, reuse_a, reuse_b)
    return __builtin_amdgcn_wmma_f32_16x16x32_bf16(false, a, false, b,
                                                   (short)0, c, false, false);
}

// ---------------------------------------------------------------------------
// Prep: convert f32 weights to bf16, pre-packed in WMMA B-fragment order.
// B fragment (32x16, bf16) lane layout (mirror of the ISA A 16x32 layout):
//   lane l : column n = l&15, k-base kb = (l>>4)*8
//   element e (0..15): k_local = kb + (e<8 ? e : e+8)
// Packed element index = ((frag*32 + lane)*16 + e)
// ---------------------------------------------------------------------------
__global__ __launch_bounds__(256)
void pack_weights_kernel(const float* __restrict__ W1,
                         const float* __restrict__ W2,
                         const float* __restrict__ W3,
                         unsigned short* __restrict__ wpack_u16) {
    int i = blockIdx.x * 256 + threadIdx.x;
    if (i >= WPACK_ELEMS) return;
    int f    = i >> 9;          // fragment id
    int r    = i & 511;
    int lane = r >> 4;
    int e    = r & 15;
    int nloc = lane & 15;
    int kb   = (lane >> 4) << 3;
    int kloc = kb + (e < 8 ? e : e + 8);

    const float* W;
    int N, kc, nt;
    if (f < FRAG_BASE_L2)      { W = W1; N = HD;      kc = f >> 3;                 nt = f & 7; }
    else if (f < FRAG_BASE_L3) { W = W2; N = HD;      int g = f - FRAG_BASE_L2; kc = g >> 3; nt = g & 7; }
    else                       { W = W3; N = OUT_DIM; int g = f - FRAG_BASE_L3; kc = g >> 2; nt = g & 3; }

    int k = kc * 32 + kloc;
    int n = nt * 16 + nloc;
    __bf16 v = (__bf16)W[(size_t)k * N + n];
    ((__bf16*)wpack_u16)[i] = v;
}

// ---------------------------------------------------------------------------
// Main fused kernel.
// ---------------------------------------------------------------------------
__global__ __launch_bounds__(256)
void bond_mlp_kernel(const float* __restrict__ sites,
                     const float* __restrict__ bonds,
                     const float* __restrict__ states,
                     const int*   __restrict__ idx1,
                     const int*   __restrict__ idx2,
                     const float* __restrict__ b1,
                     const float* __restrict__ b2,
                     const float* __restrict__ b3,
                     const unsigned short* __restrict__ wpack_u16,
                     float* __restrict__ out) {
    extern __shared__ __align__(16) char smem_raw[];
    __bf16* lds = (__bf16*)smem_raw;
    __bf16* Wl  = lds;                // packed weight fragments
    __bf16* Al  = lds + LDS_A;        // [32][224]
    __bf16* H1l = lds + LDS_H1;       // [32][128]
    __bf16* H2l = lds + LDS_H2;       // [32][128]

    const int tid  = threadIdx.x;
    const int wave = tid >> 5;        // 0..7  == N-tile id
    const int lane = tid & 31;
    const int nloc = lane & 15;
    const int half = lane >> 4;       // 0 | 1
    const int mbase = half << 3;      // C-fragment row base within a 16-row tile

    // Gather mapping: 8 threads per row, 7 float4 quads per thread.
    // quad i of thread (arow,qi) covers columns qi*4 + 32*i  -> segment is
    // compile-time known per i (0,1: site1 | 2,3: site2 | 4,5: bond | 6: state)
    const int arow = tid >> 3;        // 0..31
    const int qi   = tid & 7;         // 0..7

    // ---- Stage all packed weights into LDS (104 KB, once per block) ----
    {
        const uint4* src = (const uint4*)wpack_u16;
        uint4*       dst = (uint4*)Wl;
        constexpr int NQ = WPACK_ELEMS * 2 / 16;   // 6656 x 16B
        for (int i = tid; i < NQ; i += 256) dst[i] = src[i];
    }

    // Biases are loop-invariant: load once.
    const float bias1 = b1[wave * 16 + nloc];
    const float bias2 = b2[wave * 16 + nloc];
    const float bias3 = (wave < L3N) ? b3[wave * 16 + nloc] : 0.f;

    float4 pf[7];                     // software-prefetched gather registers

    auto do_prefetch = [&](int tt) {
        const int row0 = tt * TILE_M;
        const int b    = row0 / NE;                 // uniform within tile
        const int e    = row0 - b * NE + arow;
        const float* s1 = sites + ((size_t)b * NSITES + idx1[e]) * SITE_LEN;
        const float* s2 = sites + ((size_t)b * NSITES + idx2[e]) * SITE_LEN;
        const float* bb = bonds + ((size_t)b * NE + e) * SITE_LEN;
        const float* st = states + b * 32;
        pf[0] = *(const float4*)(s1 + qi * 4);
        pf[1] = *(const float4*)(s1 + 32 + qi * 4);
        pf[2] = *(const float4*)(s2 + qi * 4);
        pf[3] = *(const float4*)(s2 + 32 + qi * 4);
        pf[4] = *(const float4*)(bb + qi * 4);
        pf[5] = *(const float4*)(bb + 32 + qi * 4);
        pf[6] = *(const float4*)(st + qi * 4);
    };

    __syncthreads();

    int t = blockIdx.x;
    if (t < NTILES) do_prefetch(t);

    for (; t < NTILES; t += gridDim.x) {
        const int row0 = t * TILE_M;

        // ---- Commit prefetched A panel to LDS (f32 -> bf16) ----
        {
            __bf16* adst = Al + arow * IN_DIM + qi * 4;
            #pragma unroll
            for (int i = 0; i < 7; ++i) {
                float4 v = pf[i];
                *(v4bf*)(adst + 32 * i) =
                    (v4bf){ (__bf16)v.x, (__bf16)v.y, (__bf16)v.z, (__bf16)v.w };
            }
        }
        __syncthreads();

        // ---- Issue next tile's gather now; overlaps with all WMMA below ----
        {
            int tn = t + gridDim.x;
            if (tn < NTILES) do_prefetch(tn);
        }

        // ---- Layer 1: [32x224] @ [224x128] ----
        {
            v8f acc0 = {0.f,0.f,0.f,0.f,0.f,0.f,0.f,0.f};
            v8f acc1 = {0.f,0.f,0.f,0.f,0.f,0.f,0.f,0.f};
            #pragma unroll
            for (int kc = 0; kc < L1K; ++kc) {
                Frag a0, a1, bw;
                const uint4* bp = (const uint4*)(Wl + (size_t)(kc * L1N + wave) * FRAG_ELEMS + lane * 16);
                bw.q[0] = bp[0];
                bw.q[1] = bp[1];
                const uint4* ap0 = (const uint4*)(Al + nloc * IN_DIM + kc * 32 + (half << 3));
                const uint4* ap1 = (const uint4*)(Al + (16 + nloc) * IN_DIM + kc * 32 + (half << 3));
                a0.q[0] = ap0[0];  a0.q[1] = ap0[2];   // k_local [kb,kb+8) and [kb+16,kb+24)
                a1.q[0] = ap1[0];  a1.q[1] = ap1[2];
                acc0 = wmma_bf16(a0.v, bw.v, acc0);    // two independent chains
                acc1 = wmma_bf16(a1.v, bw.v, acc1);
            }
            #pragma unroll
            for (int i = 0; i < 8; ++i) {
                float h0 = fmaxf(acc0[i] + bias1, 0.f);
                float h1 = fmaxf(acc1[i] + bias1, 0.f);
                H1l[(mbase + i) * HD + wave * 16 + nloc]        = (__bf16)h0;
                H1l[(16 + mbase + i) * HD + wave * 16 + nloc]   = (__bf16)h1;
            }
        }
        __syncthreads();

        // ---- Layer 2: [32x128] @ [128x128] ----
        {
            v8f acc0 = {0.f,0.f,0.f,0.f,0.f,0.f,0.f,0.f};
            v8f acc1 = {0.f,0.f,0.f,0.f,0.f,0.f,0.f,0.f};
            #pragma unroll
            for (int kc = 0; kc < L2K; ++kc) {
                Frag a0, a1, bw;
                const uint4* bp = (const uint4*)(Wl + (size_t)(FRAG_BASE_L2 + kc * L2N + wave) * FRAG_ELEMS + lane * 16);
                bw.q[0] = bp[0];
                bw.q[1] = bp[1];
                const uint4* ap0 = (const uint4*)(H1l + nloc * HD + kc * 32 + (half << 3));
                const uint4* ap1 = (const uint4*)(H1l + (16 + nloc) * HD + kc * 32 + (half << 3));
                a0.q[0] = ap0[0];  a0.q[1] = ap0[2];
                a1.q[0] = ap1[0];  a1.q[1] = ap1[2];
                acc0 = wmma_bf16(a0.v, bw.v, acc0);
                acc1 = wmma_bf16(a1.v, bw.v, acc1);
            }
            #pragma unroll
            for (int i = 0; i < 8; ++i) {
                float h0 = fmaxf(acc0[i] + bias2, 0.f);
                float h1 = fmaxf(acc1[i] + bias2, 0.f);
                H2l[(mbase + i) * HD + wave * 16 + nloc]        = (__bf16)h0;
                H2l[(16 + mbase + i) * HD + wave * 16 + nloc]   = (__bf16)h1;
            }
        }
        __syncthreads();

        // ---- Layer 3: [32x128] @ [128x64], waves 0..3 only ----
        if (wave < L3N) {
            v8f acc0 = {0.f,0.f,0.f,0.f,0.f,0.f,0.f,0.f};
            v8f acc1 = {0.f,0.f,0.f,0.f,0.f,0.f,0.f,0.f};
            #pragma unroll
            for (int kc = 0; kc < L3K; ++kc) {
                Frag a0, a1, bw;
                const uint4* bp = (const uint4*)(Wl + (size_t)(FRAG_BASE_L3 + kc * L3N + wave) * FRAG_ELEMS + lane * 16);
                bw.q[0] = bp[0];
                bw.q[1] = bp[1];
                const uint4* ap0 = (const uint4*)(H2l + nloc * HD + kc * 32 + (half << 3));
                const uint4* ap1 = (const uint4*)(H2l + (16 + nloc) * HD + kc * 32 + (half << 3));
                a0.q[0] = ap0[0];  a0.q[1] = ap0[2];
                a1.q[0] = ap1[0];  a1.q[1] = ap1[2];
                acc0 = wmma_bf16(a0.v, bw.v, acc0);
                acc1 = wmma_bf16(a1.v, bw.v, acc1);
            }
            #pragma unroll
            for (int i = 0; i < 8; ++i) {
                int r0 = row0 + mbase + i;
                int r1 = row0 + 16 + mbase + i;
                out[(size_t)r0 * OUT_DIM + wave * 16 + nloc] = acc0[i] + bias3;
                out[(size_t)r1 * OUT_DIM + wave * 16 + nloc] = acc1[i] + bias3;
            }
        }
        // next iteration's first __syncthreads() fences Al reuse
    }
}

// ---------------------------------------------------------------------------
extern "C" void kernel_launch(void* const* d_in, const int* in_sizes, int n_in,
                              void* d_out, int out_size, void* d_ws, size_t ws_size,
                              hipStream_t stream) {
    const float* sites  = (const float*)d_in[0];
    const float* bonds  = (const float*)d_in[1];
    const float* states = (const float*)d_in[2];
    const int*   idx1   = (const int*)d_in[3];
    const int*   idx2   = (const int*)d_in[4];
    const float* W1     = (const float*)d_in[5];
    const float* b1     = (const float*)d_in[6];
    const float* W2     = (const float*)d_in[7];
    const float* b2     = (const float*)d_in[8];
    const float* W3     = (const float*)d_in[9];
    const float* b3     = (const float*)d_in[10];
    float* out = (float*)d_out;
    unsigned short* wpack = (unsigned short*)d_ws;   // bf16 packed weights

    // 1) pack weights -> bf16 WMMA fragments in workspace
    pack_weights_kernel<<<(WPACK_ELEMS + 255) / 256, 256, 0, stream>>>(W1, W2, W3, wpack);

    // 2) fused gather + 3-layer MLP
    size_t lds_bytes = (size_t)LDS_TOT_ELEMS * sizeof(unsigned short);  // 137216 B
    bond_mlp_kernel<<<2048, 256, lds_bytes, stream>>>(sites, bonds, states, idx1, idx2,
                                                      b1, b2, b3, wpack, out);
}
